// MultiHeadAttention_463856468337
// MI455X (gfx1250) — compile-verified, matrix operand order checked by reading, not run on hardware
//
#include <hip/hip_runtime.h>
#include <hip/hip_bf16.h>

#define B_   4
#define S_   1024
#define D_   1024
#define H_   16
#define HD_  64
#define SCALE_ 0.125f   // HEAD_DIM^-0.5 = 64^-0.5
#define EPS_ 1e-5f

typedef __attribute__((ext_vector_type(16))) __bf16 v16bf;
typedef __attribute__((ext_vector_type(8)))  __bf16 v8bf;
typedef __attribute__((ext_vector_type(8)))  float  v8f;
typedef __attribute__((ext_vector_type(4)))  unsigned int v4u;
typedef __attribute__((ext_vector_type(8)))  int v8i;
typedef __attribute__((ext_vector_type(4)))  int v4i;

static __device__ __forceinline__ v8bf ld8(const __bf16* p) { return *(const v8bf*)p; }
static __device__ __forceinline__ v16bf cat8(v8bf a, v8bf b) {
  return __builtin_shufflevector(a, b, 0,1,2,3,4,5,6,7,8,9,10,11,12,13,14,15);
}
// load one 16x32(K) bf16 fragment from global: two 16B chunks at {hf*8, 16+hf*8} elems
static __device__ __forceinline__ v16bf ldfrag(const __bf16* row, int hf) {
  return cat8(ld8(row + hf * 8), ld8(row + 16 + hf * 8));
}
static __device__ __forceinline__ v8f wmma_bf16(v16bf a, v16bf b, v8f c) {
  return __builtin_amdgcn_wmma_f32_16x16x32_bf16(false, a, false, b, (short)0, c, false, false);
}
// lane-group reduction helper: exchange with lane^16 (SWAPX16 ds_swizzle pattern)
static __device__ __forceinline__ float xor16(float v) {
  int i = __builtin_amdgcn_ds_swizzle(__builtin_bit_cast(int, v), 0x401f);
  return __builtin_bit_cast(float, i);
}

// ---------------------------------------------------------------- convert
__global__ void __launch_bounds__(256)
cvt_f32_bf16(const float* __restrict__ in, __bf16* __restrict__ out, int n) {
  int i = blockIdx.x * blockDim.x + threadIdx.x;
  int stride = gridDim.x * blockDim.x;
  for (; i < n; i += stride) out[i] = (__bf16)in[i];
}

// ---------------------------------------------------------------- GEMM: out[m,n] = sum_k A[m,k]*W[n,k] + bias[n]
// M=4096, N=K=1024.  Block = 8 waves sharing one 32-row A strip; the strip is DMA'd
// once into LDS by the Tensor Data Mover (tensor_load_to_lds + s_wait_tensorcnt),
// with 16B padding per 1024B to spread LDS banks.  Each wave computes a 32x64 C
// block (2x4 WMMA tiles): A frags from LDS (ds_load_b128), B frags from global.
#define LDS_ROW_STRIDE 2080   // 2048B payload + 2x16B TDM pad per row
__global__ void __launch_bounds__(256)
gemm_bf16_bias(const __bf16* __restrict__ A, const __bf16* __restrict__ W,
               const float* __restrict__ bias, float* __restrict__ out) {
  __shared__ __align__(16) unsigned char ldsA[32 * LDS_ROW_STRIDE];
  const int lane = threadIdx.x & 31;
  const int hf   = lane >> 4;       // lane half: K-interleave selector
  const int r    = lane & 15;
  const int w    = blockIdx.x * 8 + (threadIdx.x >> 5);
  const int nblk = w & 15;          // 16 column blocks of 64
  const int mblk = w >> 4;          // 128 row blocks of 32 (uniform across the block)

  if ((threadIdx.x >> 5) == 0) {
    // Tensor DMA descriptor (ISA 8.3-8.6): 2D tile 1024x32 bf16, row stride 1024 elems,
    // LDS pad 16B per 1024B payload.
    const unsigned long long ga = (unsigned long long)(const void*)(A + (long)mblk * 32 * D_);
    v4u g0;
    g0[0] = 1u;                                              // count=1, user mode
    g0[1] = 0u;                                              // lds_addr = 0 (ldsA at LDS base)
    g0[2] = (unsigned)ga;                                    // global_addr[31:0]
    g0[3] = (unsigned)((ga >> 32) & 0x1ffffffu) | (2u << 30);// global_addr[56:32] | type=2
    v8i g1;
    g1[0] = (1 << 16) | (1 << 20) | (7 << 22) | (3 << 25);   // data_size=2B, pad_en, 1024B interval, 16B pad
    g1[1] = (int)(1024u << 16);                              // tensor_dim0 = 1024
    g1[2] = (int)(32u << 16);                                // tensor_dim1 = 32
    g1[3] = (int)(1024u << 16);                              // tile_dim0 = 1024
    g1[4] = 32;                                              // tile_dim1 = 32, tile_dim2 = 0
    g1[5] = 1024;                                            // tensor_dim0_stride = 1024
    g1[6] = 0;  g1[7] = 0;
    v4i g2; g2[0] = 1; g2[1] = 1; g2[2] = 0; g2[3] = 0;      // dims 2/3 = 1 (unused)
    v4i g3; g3[0] = 0; g3[1] = (int)(1u << 16); g3[2] = 0; g3[3] = 0; // dim4 = 1
    v8i g4 = {0, 0, 0, 0, 0, 0, 0, 0};                       // unused extra descriptor group
    __builtin_amdgcn_tensor_load_to_lds(g0, g1, g2, g3, g4, 0);
    __builtin_amdgcn_s_wait_tensorcnt((short)0);
  }
  __syncthreads();
  // The TDM wrote ldsA behind the compiler's back; force it to treat the LDS
  // contents as unknown so the fragment reads below are real ds_load_b128s.
  asm volatile("" :: "v"(ldsA) : "memory");

  const unsigned aoff0 = (unsigned)r * LDS_ROW_STRIDE;
  const unsigned aoff1 = aoff0 + 16u * LDS_ROW_STRIDE;
  const __bf16* br0 = W + (long)(nblk * 64 + r) * D_;   // B column n == W row n (contiguous)
  const __bf16* br1 = br0 + 16 * D_;
  const __bf16* br2 = br0 + 32 * D_;
  const __bf16* br3 = br0 + 48 * D_;

  v8f acc[2][4] = {};
#pragma unroll 2
  for (int k0 = 0; k0 < D_; k0 += 32) {
    // LDS byte offset of this wave-half's first 16B chunk (account for mid-row pad)
    const unsigned kb = 2u * (unsigned)k0 + 16u * (unsigned)hf + ((k0 >= 512) ? 16u : 0u);
    const v16bf a0 = cat8(*(const v8bf*)(ldsA + aoff0 + kb),
                          *(const v8bf*)(ldsA + aoff0 + kb + 32));
    const v16bf a1 = cat8(*(const v8bf*)(ldsA + aoff1 + kb),
                          *(const v8bf*)(ldsA + aoff1 + kb + 32));
    const v16bf b0 = ldfrag(br0 + k0, hf);
    const v16bf b1 = ldfrag(br1 + k0, hf);
    const v16bf b2 = ldfrag(br2 + k0, hf);
    const v16bf b3 = ldfrag(br3 + k0, hf);
    acc[0][0] = wmma_bf16(a0, b0, acc[0][0]);
    acc[0][1] = wmma_bf16(a0, b1, acc[0][1]);
    acc[0][2] = wmma_bf16(a0, b2, acc[0][2]);
    acc[0][3] = wmma_bf16(a0, b3, acc[0][3]);
    acc[1][0] = wmma_bf16(a1, b0, acc[1][0]);
    acc[1][1] = wmma_bf16(a1, b1, acc[1][1]);
    acc[1][2] = wmma_bf16(a1, b2, acc[1][2]);
    acc[1][3] = wmma_bf16(a1, b3, acc[1][3]);
  }
  const int nb = nblk * 64 + r;
  const float bb[4] = {bias[nb], bias[nb + 16], bias[nb + 32], bias[nb + 48]};
#pragma unroll
  for (int mi = 0; mi < 2; ++mi) {
#pragma unroll
    for (int i = 0; i < 8; ++i) {
      float* o = out + (long)(mblk * 32 + mi * 16 + 8 * hf + i) * D_ + nb;
#pragma unroll
      for (int t = 0; t < 4; ++t) o[t * 16] = acc[mi][t][i] + bb[t];
    }
  }
}

// ---------------------------------------------------------------- LayerNorm over rows of 1024
// mode 0: fp32 row-major out.  mode 1: bf16 head-major [B,H,S,Hd].  mode 2: bf16 transposed [B,H,Hd,S].
__global__ void __launch_bounds__(256)
ln_rows(const float* __restrict__ in, const float* __restrict__ gamma,
        const float* __restrict__ beta, float scale, void* __restrict__ out, int mode) {
  __shared__ float s1[256];
  __shared__ float s2[256];
  const int row = blockIdx.x;
  const int tid = threadIdx.x;
  const float4 v = ((const float4*)(in + (long)row * D_))[tid];
  s1[tid] = v.x + v.y + v.z + v.w;
  s2[tid] = v.x * v.x + v.y * v.y + v.z * v.z + v.w * v.w;
  __syncthreads();
  for (int off = 128; off > 0; off >>= 1) {
    if (tid < off) { s1[tid] += s1[tid + off]; s2[tid] += s2[tid + off]; }
    __syncthreads();
  }
  const float mean = s1[0] * (1.0f / D_);
  const float var  = s2[0] * (1.0f / D_) - mean * mean;
  const float rstd = rsqrtf(var + EPS_);
  const int c0 = tid * 4;
  float y[4] = {v.x, v.y, v.z, v.w};
#pragma unroll
  for (int j = 0; j < 4; ++j) {
    const int c = c0 + j;
    y[j] = ((y[j] - mean) * rstd * gamma[c] + beta[c]) * scale;
  }
  if (mode == 0) {
    float* o = (float*)out + (long)row * D_ + c0;
#pragma unroll
    for (int j = 0; j < 4; ++j) o[j] = y[j];
  } else {
    const int b = row >> 10, s = row & 1023;
    __bf16* o = (__bf16*)out;
#pragma unroll
    for (int j = 0; j < 4; ++j) {
      const int c = c0 + j, h = c >> 6, d = c & 63;
      const long idx = (mode == 1) ? ((long)((b * H_ + h) * S_ + s) * HD_ + d)
                                   : ((long)((b * H_ + h) * HD_ + d) * S_ + s);
      o[idx] = (__bf16)y[j];
    }
  }
}

// ---------------------------------------------------------------- fused flash attention
// One wave = 32 queries (two 16-q groups) of one (b,h); streams all keys in 32-steps.
// S^T = K*Q^T so softmax reduces in-lane + lane^16, and P^T lands directly in the
// B-operand layout of O^T += V^T * P^T.  K/V^T fragments are shared by both q groups.
__global__ void __launch_bounds__(128)
attn_fused(const __bf16* __restrict__ Q, const __bf16* __restrict__ K,
           const __bf16* __restrict__ Vt, __bf16* __restrict__ out) {
  const int lane = threadIdx.x & 31;
  const int hf   = lane >> 4;
  const int r    = lane & 15;
  const int wid  = blockIdx.x * 4 + (threadIdx.x >> 5);
  const int qt   = wid & (S_ / 32 - 1);        // 32 query blocks of 32
  const int bh   = wid >> 5;
  const int b    = bh >> 4, h = bh & 15;
  const __bf16* Qp = Q  + (long)bh * S_ * HD_;
  const __bf16* Kp = K  + (long)bh * S_ * HD_;
  const __bf16* Vp = Vt + (long)bh * HD_ * S_;
  const int qbase = qt * 32;

  // Q^T B-fragments, two groups of 16 query columns; live across whole loop
  const __bf16* qr0 = Qp + (long)(qbase + r) * HD_;
  const __bf16* qr1 = Qp + (long)(qbase + 16 + r) * HD_;
  const v16bf qA0 = ldfrag(qr0, hf),      qA1 = ldfrag(qr0 + 32, hf);
  const v16bf qB0 = ldfrag(qr1, hf),      qB1 = ldfrag(qr1 + 32, hf);

  float mA = -3.0e38f, lA = 0.0f;
  float mB = -3.0e38f, lB = 0.0f;
  v8f oA[4] = {}, oB[4] = {};   // O^T tiles: d = t*16 + i + 8*hf, q = r (per group)

  for (int kt = 0; kt < S_; kt += 32) {
    // shared K A-fragments (rows = keys, contiguous along d)
    const __bf16* ka = Kp + (long)(kt + r) * HD_;
    const __bf16* kb = Kp + (long)(kt + 16 + r) * HD_;
    const v16bf ka0 = ldfrag(ka, hf),      ka1 = ldfrag(ka + 32, hf);
    const v16bf kb0 = ldfrag(kb, hf),      kb1 = ldfrag(kb + 32, hf);

    // scores^T for both q groups (keys kt..kt+15 in c0/c2, kt+16..kt+31 in c1/c3)
    v8f c0 = {}, c1 = {}, c2 = {}, c3 = {};
    c0 = wmma_bf16(ka0, qA0, c0);  c0 = wmma_bf16(ka1, qA1, c0);
    c1 = wmma_bf16(kb0, qA0, c1);  c1 = wmma_bf16(kb1, qA1, c1);
    c2 = wmma_bf16(ka0, qB0, c2);  c2 = wmma_bf16(ka1, qB1, c2);
    c3 = wmma_bf16(kb0, qB0, c3);  c3 = wmma_bf16(kb1, qB1, c3);

    // ---- group A softmax update ----
    float tm = -3.0e38f;
#pragma unroll
    for (int j = 0; j < 8; ++j) {
      c0[j] = fminf(10.0f, fmaxf(-10.0f, c0[j]));
      c1[j] = fminf(10.0f, fmaxf(-10.0f, c1[j]));
      tm = fmaxf(tm, fmaxf(c0[j], c1[j]));
    }
    tm = fmaxf(tm, xor16(tm));
    float mn = fmaxf(mA, tm);
    float alpha = __expf(mA - mn);
    float ts = 0.0f;
    v16bf pA;
#pragma unroll
    for (int j = 0; j < 8; ++j) {
      const float p0 = __expf(c0[j] - mn); ts += p0; pA[j]     = (__bf16)p0;
      const float p1 = __expf(c1[j] - mn); ts += p1; pA[8 + j] = (__bf16)p1;
    }
    ts += xor16(ts);
    lA = lA * alpha + ts;  mA = mn;
#pragma unroll
    for (int t = 0; t < 4; ++t)
#pragma unroll
      for (int j = 0; j < 8; ++j) oA[t][j] *= alpha;

    // ---- group B softmax update ----
    tm = -3.0e38f;
#pragma unroll
    for (int j = 0; j < 8; ++j) {
      c2[j] = fminf(10.0f, fmaxf(-10.0f, c2[j]));
      c3[j] = fminf(10.0f, fmaxf(-10.0f, c3[j]));
      tm = fmaxf(tm, fmaxf(c2[j], c3[j]));
    }
    tm = fmaxf(tm, xor16(tm));
    mn = fmaxf(mB, tm);
    alpha = __expf(mB - mn);
    ts = 0.0f;
    v16bf pB;
#pragma unroll
    for (int j = 0; j < 8; ++j) {
      const float p0 = __expf(c2[j] - mn); ts += p0; pB[j]     = (__bf16)p0;
      const float p1 = __expf(c3[j] - mn); ts += p1; pB[8 + j] = (__bf16)p1;
    }
    ts += xor16(ts);
    lB = lB * alpha + ts;  mB = mn;
#pragma unroll
    for (int t = 0; t < 4; ++t)
#pragma unroll
      for (int j = 0; j < 8; ++j) oB[t][j] *= alpha;

    // ---- O^T += V^T * P^T, V^T fragments shared by both groups ----
#pragma unroll
    for (int t = 0; t < 4; ++t) {
      const __bf16* vr = Vp + (long)(t * 16 + r) * S_ + kt;
      const v16bf va = ldfrag(vr, hf);
      oA[t] = wmma_bf16(va, pA, oA[t]);
      oB[t] = wmma_bf16(va, pB, oB[t]);
    }
  }

  const float invA = 1.0f / lA;
  const float invB = 1.0f / lB;
  // store to attention output in row-major [B*S, D] (input layout of the Wo GEMM)
  __bf16* orA = out + (long)(b * S_ + qbase + r) * D_ + h * HD_ + 8 * hf;
  __bf16* orB = out + (long)(b * S_ + qbase + 16 + r) * D_ + h * HD_ + 8 * hf;
#pragma unroll
  for (int t = 0; t < 4; ++t) {
#pragma unroll
    for (int i = 0; i < 8; ++i) {
      orA[t * 16 + i] = (__bf16)(oA[t][i] * invA);
      orB[t * 16 + i] = (__bf16)(oB[t][i] * invB);
    }
  }
}

// ---------------------------------------------------------------- launch
extern "C" void kernel_launch(void* const* d_in, const int* in_sizes, int n_in,
                              void* d_out, int out_size, void* d_ws, size_t ws_size,
                              hipStream_t stream) {
  const float* x  = (const float*)d_in[0];
  const float* Wq = (const float*)d_in[1];  const float* bq = (const float*)d_in[2];
  const float* Wk = (const float*)d_in[3];  const float* bk = (const float*)d_in[4];
  const float* Wv = (const float*)d_in[5];  const float* bv = (const float*)d_in[6];
  const float* Wo = (const float*)d_in[7];  const float* bo = (const float*)d_in[8];
  const float* qg = (const float*)d_in[9];  const float* qbeta = (const float*)d_in[10];
  const float* kg = (const float*)d_in[11]; const float* kbeta = (const float*)d_in[12];
  const float* vg = (const float*)d_in[13]; const float* vbeta = (const float*)d_in[14];
  const float* og = (const float*)d_in[15]; const float* obeta = (const float*)d_in[16];

  char* ws = (char*)d_ws;
  __bf16* xb  = (__bf16*)(ws);                // 8 MB: x in bf16
  __bf16* wqb = (__bf16*)(ws + (8  << 20));   // 2 MB each: weights in bf16
  __bf16* wkb = (__bf16*)(ws + (10 << 20));
  __bf16* wvb = (__bf16*)(ws + (12 << 20));
  __bf16* wob = (__bf16*)(ws + (14 << 20));
  float*  tmp = (float*) (ws + (16 << 20));   // 16 MB: pre-LN GEMM result
  __bf16* qh  = (__bf16*)(ws + (32 << 20));   // 8 MB: Q [B,H,S,Hd] bf16 (scaled)
  __bf16* kh  = (__bf16*)(ws + (40 << 20));   // 8 MB: K [B,H,S,Hd] bf16
  __bf16* vt  = (__bf16*)(ws + (48 << 20));   // 8 MB: V^T [B,H,Hd,S] bf16
  __bf16* ab  = (__bf16*)(ws + (56 << 20));   // 8 MB: attention out, row-major bf16

  cvt_f32_bf16<<<2048, 256, 0, stream>>>(x,  xb,  B_ * S_ * D_);
  cvt_f32_bf16<<<512,  256, 0, stream>>>(Wq, wqb, D_ * D_);
  cvt_f32_bf16<<<512,  256, 0, stream>>>(Wk, wkb, D_ * D_);
  cvt_f32_bf16<<<512,  256, 0, stream>>>(Wv, wvb, D_ * D_);
  cvt_f32_bf16<<<512,  256, 0, stream>>>(Wo, wob, D_ * D_);

  gemm_bf16_bias<<<256, 256, 0, stream>>>(xb, wqb, bq, tmp);
  ln_rows<<<4096, 256, 0, stream>>>(tmp, qg, qbeta, SCALE_, (void*)qh, 1);
  gemm_bf16_bias<<<256, 256, 0, stream>>>(xb, wkb, bk, tmp);
  ln_rows<<<4096, 256, 0, stream>>>(tmp, kg, kbeta, 1.0f, (void*)kh, 1);
  gemm_bf16_bias<<<256, 256, 0, stream>>>(xb, wvb, bv, tmp);
  ln_rows<<<4096, 256, 0, stream>>>(tmp, vg, vbeta, 1.0f, (void*)vt, 2);

  attn_fused<<<512, 128, 0, stream>>>(qh, kh, vt, ab);

  gemm_bf16_bias<<<256, 256, 0, stream>>>(ab, wob, bo, tmp);
  ln_rows<<<4096, 256, 0, stream>>>(tmp, og, obeta, 1.0f, d_out, 0);
}
